// PAM_Use_2422361555677
// MI455X (gfx1250) — matrix operand order repaired and checked by reading, not compiled, not run on hardware
//
#include <hip/hip_runtime.h>

// ---------------------------------------------------------------------------
// PAM (position attention) for MI455X / gfx1250, wave32 + WMMA bf16.
// B=4, C=512, H=W=64 -> N=4096, D=64.
//
// Pipeline:
//   cvt_bf16   : value_w / key_w  f32 -> bf16 (one-time)
//   transpose_x: x [B][C][N] f32 -> xT [B][N][C] bf16 (one-time, LDS tiles)
//   proj_gemm  : V = wv*x + b -> bf16 [B][C][N] ; K = wk*x + b -> bf16 [B][N][D]
//   pam_attn   : flash-style fused energy -> softmax -> V*attn^T
//                K tiles double-buffered into LDS via global_load_async_to_lds,
//                V prefetched a round ahead (global_prefetch_b8),
//                out = gamma*O/rowsum + x (f32)
// ---------------------------------------------------------------------------

typedef __attribute__((ext_vector_type(16))) __bf16 v16bf;
typedef __attribute__((ext_vector_type(8)))  float  v8f;

#define BATCH 4
#define CCH   512
#define NPOS  4096
#define DDIM  64
#define NROUNDS (NPOS / 256)

__device__ __forceinline__ unsigned short bf_bits(float f) {
    union { float f; unsigned u; } a; a.f = f;
    unsigned r = a.u + 0x7FFFu + ((a.u >> 16) & 1u);   // round-to-nearest-even
    return (unsigned short)(r >> 16);
}

// ---------------------------------------------------------------------------
// f32 -> bf16 elementwise, 8 elements per thread (2x float4 in, 1x uint4 out)
// ---------------------------------------------------------------------------
__global__ __launch_bounds__(256) void cvt_bf16_kernel(
    const float* __restrict__ src, unsigned short* __restrict__ dst, int n8)
{
    const int t = blockIdx.x * blockDim.x + threadIdx.x;
    if (t >= n8) return;
    const float4 a = ((const float4*)src)[t * 2 + 0];
    const float4 b = ((const float4*)src)[t * 2 + 1];
    unsigned short v[8];
    v[0] = bf_bits(a.x); v[1] = bf_bits(a.y); v[2] = bf_bits(a.z); v[3] = bf_bits(a.w);
    v[4] = bf_bits(b.x); v[5] = bf_bits(b.y); v[6] = bf_bits(b.z); v[7] = bf_bits(b.w);
    ((uint4*)dst)[t] = *(const uint4*)v;
}

// ---------------------------------------------------------------------------
// x [B][C][N] f32 -> xT [B][N][C] bf16, 64x64 tiles through LDS.
// ---------------------------------------------------------------------------
__global__ __launch_bounds__(256) void transpose_x_kernel(
    const float* __restrict__ x, unsigned short* __restrict__ xT)
{
    __shared__ unsigned short tile[64][72];   // [n][c], pad -> 144B rows (16B mult)

    const int t  = threadIdx.x;
    const int n0 = blockIdx.x * 64;
    const int c0 = blockIdx.y * 64;
    const int b  = blockIdx.z;

    #pragma unroll
    for (int i = 0; i < 4; ++i) {
        const int fidx = t + 256 * i;
        const int row  = fidx >> 4;          // c within tile
        const int c4   = fidx & 15;          // float4 index along n
        const float4 v = *(const float4*)(x + ((size_t)b * CCH + c0 + row) * NPOS
                                            + n0 + c4 * 4);
        tile[c4 * 4 + 0][row] = bf_bits(v.x);
        tile[c4 * 4 + 1][row] = bf_bits(v.y);
        tile[c4 * 4 + 2][row] = bf_bits(v.z);
        tile[c4 * 4 + 3][row] = bf_bits(v.w);
    }
    __syncthreads();

    #pragma unroll
    for (int i = 0; i < 2; ++i) {
        const int sidx = t + 256 * i;
        const int n = sidx >> 3;
        const int g = sidx & 7;
        const uint4 v = *(const uint4*)&tile[n][g * 8];
        *(uint4*)(xT + ((size_t)b * NPOS + n0 + n) * CCH + c0 + g * 8) = v;
    }
}

// ---------------------------------------------------------------------------
// proj GEMM: out[m,n] = sum_c w[m,c]*xT[n,c] + bias[m], bf16 in, bf16 out.
// One wave per block computes a 16(m) x 64(n) tile, K-loop step 32.
// mode 0: out stored [B][M][N]   (value projection)
// mode 1: out stored [B][N][M]   (key projection, transposed for attn B-frags)
// ---------------------------------------------------------------------------
__global__ __launch_bounds__(32) void proj_gemm_kernel(
    const unsigned short* __restrict__ xT,   // [B][N][Cin] bf16
    const unsigned short* __restrict__ w,    // [M][Cin] bf16
    const float* __restrict__ bias,          // [M]
    unsigned short* __restrict__ out,
    int Cin, int Nn, int M, int mode)
{
    const int lane = threadIdx.x & 31;
    const int col  = lane & 15;
    const int h    = lane >> 4;
    const int n0   = blockIdx.x * 64;
    const int m0   = blockIdx.y * 16;
    const int b    = blockIdx.z;

    v8f acc[4] = {};

    #pragma unroll 4
    for (int c0 = 0; c0 < Cin; c0 += 32) {
        union { v16bf v; uint4 u[2]; } a;
        const unsigned short* wp = w + (size_t)(m0 + col) * Cin + c0;
        a.u[0] = *(const uint4*)(wp + h * 16);
        a.u[1] = *(const uint4*)(wp + h * 16 + 8);

        #pragma unroll
        for (int j = 0; j < 4; ++j) {
            const unsigned short* xp =
                xT + ((size_t)b * Nn + n0 + 16 * j + col) * Cin + c0 + h * 16;
            const v16bf bb = *(const v16bf*)xp;
            acc[j] = __builtin_amdgcn_wmma_f32_16x16x32_bf16(
                false, a.v, false, bb, (short)0, acc[j], false, false);
        }
    }

    if (mode == 0) {
        #pragma unroll
        for (int j = 0; j < 4; ++j) {
            #pragma unroll
            for (int r = 0; r < 8; ++r) {
                const int mm = m0 + 8 * h + r;
                out[((size_t)b * M + mm) * Nn + n0 + 16 * j + col] =
                    bf_bits(acc[j][r] + bias[mm]);
            }
        }
    } else {
        #pragma unroll
        for (int j = 0; j < 4; ++j) {
            unsigned short vals[8];
            #pragma unroll
            for (int r = 0; r < 8; ++r) {
                const int mm = m0 + 8 * h + r;
                vals[r] = bf_bits(acc[j][r] + bias[mm]);
            }
            unsigned short* p = out + ((size_t)b * Nn + n0 + 16 * j + col) * M
                                    + m0 + 8 * h;
            *(uint4*)p = *(const uint4*)vals;
        }
    }
}

// ---------------------------------------------------------------------------
// Fused flash attention + output.
// Block = 256 threads = 8 waves; one 16-query tile per block.
// Wave w: computes S for key chunk [round*256 + 32w, +32), accumulates O for
// channel strip [64w, 64w+64). K tiles are double-buffered into LDS with
// global_load_async_to_lds_b128 (issued a round ahead, drained with
// s_wait_asynccnt at the round boundary). V is prefetched a round ahead.
// ---------------------------------------------------------------------------
__global__ __launch_bounds__(256) void pam_attn_kernel(
    const float* __restrict__ q,            // [B][N][D] f32
    const unsigned short* __restrict__ Kp,  // [B][N][D] bf16
    const unsigned short* __restrict__ Vp,  // [B][C][N] bf16
    const float* __restrict__ x,            // [B][C][N] f32
    const float* __restrict__ gamma,        // [1]
    float* __restrict__ out)                // [B][C][N] f32
{
    __shared__ unsigned short Kstage[2][256][DDIM]; // 2 x 32KB double buffer
    __shared__ unsigned short Pld[16][256];         // P tile bf16
    __shared__ float smax[8][16];
    __shared__ float ssum[8][16];

    const int tid  = threadIdx.x;
    const int w    = tid >> 5;
    const int lane = tid & 31;
    const int col  = lane & 15;
    const int h    = lane >> 4;
    const int m0   = blockIdx.x * 16;       // query tile base
    const int b    = blockIdx.y;
    const int c_base = w * 64;              // this wave's channel strip

    // async stage of 256 K rows (keys kb..kb+255) into Kstage[nbuf]
    const unsigned short* kbase = Kp + (size_t)b * NPOS * DDIM;
    auto stage_keys = [&](int kb, int nbuf) {
        const int key = tid >> 3;           // 0..31
        const int seg = (tid & 7) * 8;      // bf16 element offset (16B chunks)
        #pragma unroll
        for (int i = 0; i < 8; ++i) {
            const unsigned short* g = kbase + (size_t)(kb + key + 32 * i) * DDIM + seg;
            const unsigned loff =
                (unsigned)(size_t)&Kstage[nbuf][key + 32 * i][seg];
            asm volatile("global_load_async_to_lds_b128 %0, %1, off"
                         :: "v"(loff), "v"(g) : "memory");
        }
    };

    // Q A-fragments (16 queries x 64 D = 2 K-chunks): vector f32 loads + cvt.
    v16bf qa[2];
    #pragma unroll
    for (int kk = 0; kk < 2; ++kk) {
        const float* qp = q + ((size_t)b * NPOS + m0 + col) * DDIM + kk * 32;
        const float4 lo0 = *(const float4*)(qp + h * 16);
        const float4 lo1 = *(const float4*)(qp + h * 16 + 4);
        const float4 hi0 = *(const float4*)(qp + h * 16 + 8);
        const float4 hi1 = *(const float4*)(qp + h * 16 + 12);
        unsigned short v[16];
        v[0]=bf_bits(lo0.x); v[1]=bf_bits(lo0.y); v[2]=bf_bits(lo0.z); v[3]=bf_bits(lo0.w);
        v[4]=bf_bits(lo1.x); v[5]=bf_bits(lo1.y); v[6]=bf_bits(lo1.z); v[7]=bf_bits(lo1.w);
        v[8]=bf_bits(hi0.x); v[9]=bf_bits(hi0.y); v[10]=bf_bits(hi0.z); v[11]=bf_bits(hi0.w);
        v[12]=bf_bits(hi1.x); v[13]=bf_bits(hi1.y); v[14]=bf_bits(hi1.z); v[15]=bf_bits(hi1.w);
        qa[kk] = *(const v16bf*)v;
    }

    v8f acc[4] = {};
    float runmax[8], runsum[8];
    #pragma unroll
    for (int r = 0; r < 8; ++r) { runmax[r] = -__builtin_inff(); runsum[r] = 0.f; }

    stage_keys(0, 0);   // prologue: stage round 0

    for (int round = 0; round < NROUNDS; ++round) {
        const int buf = round & 1;
        const int keybase = round * 256;

        // drain this round's K stage; also fences Pld reuse across rounds
        asm volatile("s_wait_asynccnt 0x0" ::: "memory");
        __syncthreads();

        // ---- S = Q * K^T for this wave's 32 keys (from LDS stage) ----
        v8f s[2] = {};
        #pragma unroll
        for (int jf = 0; jf < 2; ++jf) {
            #pragma unroll
            for (int kk = 0; kk < 2; ++kk) {
                const unsigned short* kp =
                    &Kstage[buf][w * 32 + 16 * jf + col][kk * 32 + h * 16];
                const v16bf kb = *(const v16bf*)kp;     // 32B LDS read
                s[jf] = __builtin_amdgcn_wmma_f32_16x16x32_bf16(
                    false, qa[kk], false, kb, (short)0, s[jf], false, false);
            }
        }

        // ---- per-wave row max over 32 keys (cols across 16 lanes) ----
        float rm[8];
        #pragma unroll
        for (int r = 0; r < 8; ++r) {
            float v = fmaxf(s[0][r], s[1][r]);
            #pragma unroll
            for (int mk = 1; mk < 16; mk <<= 1)
                v = fmaxf(v, __shfl_xor(v, mk, 32));
            rm[r] = v;
        }
        if (col == 0) {
            #pragma unroll
            for (int r = 0; r < 8; ++r) smax[w][8 * h + r] = rm[r];
        }
        __syncthreads();

        // kick off next round's K stage (overlaps softmax + O GEMM)
        if (round + 1 < NROUNDS) stage_keys(keybase + 256, buf ^ 1);

        // ---- global new max + rescale factor (replicated per wave) ----
        float nm[8], scale[8];
        #pragma unroll
        for (int r = 0; r < 8; ++r) {
            float mx = runmax[r];
            #pragma unroll
            for (int w2 = 0; w2 < 8; ++w2) mx = fmaxf(mx, smax[w2][8 * h + r]);
            nm[r]     = mx;
            scale[r]  = __expf(runmax[r] - mx);
            runmax[r] = mx;
        }

        // ---- P = exp(S - max) -> LDS (bf16), partial row sums ----
        float rs[8];
        #pragma unroll
        for (int r = 0; r < 8; ++r) {
            const float p0 = __expf(s[0][r] - nm[r]);
            const float p1 = __expf(s[1][r] - nm[r]);
            Pld[8 * h + r][w * 32 + col]      = bf_bits(p0);
            Pld[8 * h + r][w * 32 + 16 + col] = bf_bits(p1);
            float tsum = p0 + p1;
            #pragma unroll
            for (int mk = 1; mk < 16; mk <<= 1)
                tsum += __shfl_xor(tsum, mk, 32);
            rs[r] = tsum;
        }
        if (col == 0) {
            #pragma unroll
            for (int r = 0; r < 8; ++r) ssum[w][8 * h + r] = rs[r];
        }
        __syncthreads();

        // ---- update running sum, rescale O accumulators ----
        #pragma unroll
        for (int r = 0; r < 8; ++r) {
            float su = 0.f;
            #pragma unroll
            for (int w2 = 0; w2 < 8; ++w2) su += ssum[w2][8 * h + r];
            runsum[r] = runsum[r] * scale[r] + su;
            #pragma unroll
            for (int j = 0; j < 4; ++j) acc[j][r] *= scale[r];
        }

        // ---- O += P * V^T over all 8 key chunks, this wave's channels ----
        #pragma unroll
        for (int qc = 0; qc < 8; ++qc) {
            union { v16bf v; uint4 u[2]; } pa;           // A frag from LDS
            const unsigned short* pl = &Pld[col][qc * 32 + h * 16];
            pa.u[0] = *(const uint4*)pl;
            pa.u[1] = *(const uint4*)(pl + 8);
            #pragma unroll
            for (int j = 0; j < 4; ++j) {
                const unsigned short* vp =
                    Vp + ((size_t)b * CCH + c_base + 16 * j + col) * NPOS
                       + keybase + qc * 32 + h * 16;     // 32B contiguous
                const v16bf vb = *(const v16bf*)vp;
                // prefetch next round's V strip (128B line granularity)
                if ((qc & 1) == 0 && round + 1 < NROUNDS)
                    __builtin_prefetch(vp + 256, 0, 0);
                acc[j] = __builtin_amdgcn_wmma_f32_16x16x32_bf16(
                    false, pa.v, false, vb, (short)0, acc[j], false, false);
            }
        }
        // (no trailing barrier: next round's asynccnt drain + barrier covers it)
    }

    // ---- epilogue: out = gamma * O / rowsum + x ----
    const float g = gamma[0];
    float inv[8];
    #pragma unroll
    for (int r = 0; r < 8; ++r) inv[r] = g / runsum[r];

    #pragma unroll
    for (int j = 0; j < 4; ++j) {
        const int c = c_base + 16 * j + col;
        const size_t base = ((size_t)b * CCH + c) * NPOS + m0 + 8 * h;
        const float4 x0 = *(const float4*)(x + base);
        const float4 x1 = *(const float4*)(x + base + 4);
        float4 o0, o1;
        o0.x = acc[j][0] * inv[0] + x0.x;
        o0.y = acc[j][1] * inv[1] + x0.y;
        o0.z = acc[j][2] * inv[2] + x0.z;
        o0.w = acc[j][3] * inv[3] + x0.w;
        o1.x = acc[j][4] * inv[4] + x1.x;
        o1.y = acc[j][5] * inv[5] + x1.y;
        o1.z = acc[j][6] * inv[6] + x1.z;
        o1.w = acc[j][7] * inv[7] + x1.w;
        *(float4*)(out + base)     = o0;
        *(float4*)(out + base + 4) = o1;
    }
}

// ---------------------------------------------------------------------------
extern "C" void kernel_launch(void* const* d_in, const int* in_sizes, int n_in,
                              void* d_out, int out_size, void* d_ws, size_t ws_size,
                              hipStream_t stream) {
    const float* x       = (const float*)d_in[0];   // [B,C,H,W]
    const float* q       = (const float*)d_in[1];   // [B,N,D]
    const float* value_w = (const float*)d_in[2];   // [C,C]
    const float* value_b = (const float*)d_in[3];   // [C]
    const float* key_w   = (const float*)d_in[4];   // [D,C]
    const float* key_b   = (const float*)d_in[5];   // [D]
    const float* gamma   = (const float*)d_in[6];   // [1]
    float* out = (float*)d_out;

    // workspace layout (bf16 elements)
    unsigned short* Vp = (unsigned short*)d_ws;                    // [B][C][N]
    unsigned short* Kp = Vp + (size_t)BATCH * CCH * NPOS;          // [B][N][D]
    unsigned short* xT = Kp + (size_t)BATCH * NPOS * DDIM;         // [B][N][C]
    unsigned short* wv = xT + (size_t)BATCH * NPOS * CCH;          // [C][C]
    unsigned short* wk = wv + (size_t)CCH * CCH;                   // [D][C]

    {
        const int n8v = CCH * CCH / 8;
        cvt_bf16_kernel<<<dim3((n8v + 255) / 256), 256, 0, stream>>>(value_w, wv, n8v);
        const int n8k = DDIM * CCH / 8;
        cvt_bf16_kernel<<<dim3((n8k + 255) / 256), 256, 0, stream>>>(key_w, wk, n8k);
    }
    {
        dim3 gt(NPOS / 64, CCH / 64, BATCH);
        transpose_x_kernel<<<gt, 256, 0, stream>>>(x, xT);
    }
    {
        dim3 gv(NPOS / 64, CCH / 16, BATCH);
        proj_gemm_kernel<<<gv, 32, 0, stream>>>(xT, wv, value_b, Vp,
                                                CCH, NPOS, CCH, 0);
    }
    {
        dim3 gk(NPOS / 64, DDIM / 16, BATCH);
        proj_gemm_kernel<<<gk, 32, 0, stream>>>(xT, wk, key_b, Kp,
                                                CCH, NPOS, DDIM, 1);
    }
    {
        dim3 ga(NPOS / 16, BATCH);
        pam_attn_kernel<<<ga, 256, 0, stream>>>(q, Kp, Vp, x, gamma, out);
    }
}
